// GPT_1236950581814
// MI455X (gfx1250) — matrix operand order repaired and checked by reading, not compiled, not run on hardware
//
#include <hip/hip_runtime.h>
#include <stdint.h>

// GPT-2 forward (B=2,T=1024,C=1024,H=16,L=12,V=50257) for gfx1250.
// GEMMs: V_WMMA_F32_16X16X32_BF16, double-buffered LDS tiles staged by the
// Tensor Data Mover (TENSOR_LOAD_TO_LDS, TENSORcnt) with hardware OOB
// zero-fill; fallback: GLOBAL_LOAD_ASYNC_TO_LDS_B128 (ASYNCcnt).

#define B_   2
#define T_   1024
#define C_   1024
#define H_   16
#define L_   12
#define V_   50257
#define HD_  64
#define BT_  (B_ * T_)
#define BH_  (B_ * H_)
#define VPAD 50264  // V padded to multiple of 8 elems (16B) for aligned b128 fills

typedef __attribute__((ext_vector_type(16))) __bf16 v16bf;
typedef __attribute__((ext_vector_type(8)))  float  v8f;
typedef __attribute__((ext_vector_type(4)))  unsigned int v4u;
typedef __attribute__((ext_vector_type(8)))  int v8i;
typedef __attribute__((ext_vector_type(4)))  int v4i;

struct FalseC { static constexpr bool value = false; };
struct TrueC  { static constexpr bool value = true;  };

#if defined(__gfx1250__) && __has_builtin(__builtin_amdgcn_tensor_load_to_lds)
#define HAVE_TDM 1
#else
#define HAVE_TDM 0
#endif

__device__ __forceinline__ unsigned short f2bf(float f) {
  unsigned int u = __float_as_uint(f);
  unsigned int r = u + 0x7FFFu + ((u >> 16) & 1u);  // round-to-nearest-even
  return (unsigned short)(r >> 16);
}

// ---- CDNA5 async global->LDS staging (fallback path) ----------------------
__device__ __forceinline__ void async_b128(const void* lds_dst, const void* gsrc) {
  asm volatile("global_load_async_to_lds_b128 %0, %1, off"
               :
               : "v"((unsigned int)(uintptr_t)lds_dst),
                 "v"((unsigned long long)(uintptr_t)gsrc)
               : "memory");
}
__device__ __forceinline__ void wait_async0() {
#if __has_builtin(__builtin_amdgcn_s_wait_asynccnt)
  __builtin_amdgcn_s_wait_asynccnt((short)0);
#else
  asm volatile("s_wait_asynccnt 0" ::: "memory");
#endif
}
__device__ __forceinline__ void wait_tensor0() {
#if __has_builtin(__builtin_amdgcn_s_wait_tensorcnt)
  __builtin_amdgcn_s_wait_tensorcnt((short)0);
#else
  asm volatile("s_wait_tensorcnt 0" ::: "memory");
#endif
}

#if HAVE_TDM
// ---- Tensor Data Mover: 2D tile DMA (global -> LDS), D# per ISA ch.8 ------
// group0: [1:0]=count=1, [63:32]=lds_addr, [120:64]=global_addr, [127:126]=type=2
// group1: [17:16]=data_size(1 -> 2B), [79:48]=tensor_dim0 (remaining row len),
//         [111:80]=tensor_dim1 (remaining rows), [127:112]=tile_dim0,
//         [143:128]=tile_dim1, [207:160]=tensor_dim0_stride (elems)
// OOB reads return zero -> hardware zero-fill of ragged tiles.
__device__ __forceinline__ void tdm_load_2d(const void* lds_dst, const void* gsrc,
                                            unsigned tensor_d0, unsigned tensor_d1,
                                            unsigned long long stride_elems,
                                            unsigned tile_d0, unsigned tile_d1) {
  union { unsigned long long u64[2]; v4u v; } g0;
  union { unsigned long long u64[4]; v8i v; } g1;
  const unsigned long long ga  = (unsigned long long)(uintptr_t)gsrc;
  const unsigned long long lds = (unsigned int)(uintptr_t)lds_dst;
  g0.u64[0] = 1ull | (lds << 32);
  g0.u64[1] = (ga & 0x01FFFFFFFFFFFFFFull) | (2ull << 62);
  g1.u64[0] = (1ull << 16) | (((unsigned long long)(tensor_d0 & 0xFFFFu)) << 48);
  g1.u64[1] = (((unsigned long long)tensor_d0 >> 16) & 0xFFFFull)
            | ((unsigned long long)tensor_d1 << 16)
            | ((unsigned long long)tile_d0 << 48);
  g1.u64[2] = (unsigned long long)tile_d1
            | ((stride_elems & 0xFFFFFFFFull) << 32);
  g1.u64[3] = (stride_elems >> 32) & 0xFFFFull;   // dim1_stride = 0 (2D)
  v4i z4 = {0, 0, 0, 0};
#if __clang_major__ >= 23
  v8i z8 = {0, 0, 0, 0, 0, 0, 0, 0};
  __builtin_amdgcn_tensor_load_to_lds(g0.v, g1.v, z4, z4, z8, 0);
#else
  __builtin_amdgcn_tensor_load_to_lds(g0.v, g1.v, z4, z4, 0);
#endif
}
#endif  // HAVE_TDM

// ---------------------------------------------------------------------------
// Tiled bf16 WMMA GEMM:  C[M,N] = act( alpha * A[M,K] @ B[K,N] + bias ) (+resid)
// Block = 256 threads (8 waves), tile 128x128, BK=32, double-buffered LDS.
// Wave w: rows (w&3)*32, cols (w>>2)*64 -> 2x4 grid of 16x16 WMMA accums.
// Requires: M % 128 == 0, K % 32 == 0; N may be ragged. ldb % 8 == 0.
// ---------------------------------------------------------------------------
template <bool GELU, bool RESID, bool OUTBF>
__global__ __launch_bounds__(256)
void gpt_gemm_wmma_bf16(const unsigned short* __restrict__ A, int lda, long long strideA,
                        const unsigned short* __restrict__ Bm, int ldb, long long strideB,
                        const float* __restrict__ bias,
                        const float* __restrict__ resid,
                        float* __restrict__ Cf, unsigned short* __restrict__ Cb,
                        int ldc, long long strideC,
                        int M, int N, int K, float alpha, int causal_skip) {
  __shared__ __align__(16) unsigned short As[2 * 128 * 32];
  __shared__ __align__(16) unsigned short Bs[2 * 32 * 128];

  const int m0 = blockIdx.y * 128;
  const int n0 = blockIdx.x * 128;
  if (causal_skip && n0 > m0 + 127) return;  // fully-masked causal tile

  const int z = blockIdx.z;
  A  += (long long)z * strideA;
  Bm += (long long)z * strideB;
  if (Cf) Cf += (long long)z * strideC;
  if (Cb) Cb += (long long)z * strideC;

  const int tid  = threadIdx.x;
  const int lane = tid & 31;
  const int wave = tid >> 5;
  const int wm0  = (wave & 3) * 32;
  const int wn0  = (wave >> 2) * 64;
  const int hf   = lane >> 4;   // lane half -> K sub-chunk select (ISA A layout)
  const int l15  = lane & 15;

  union FragU { v16bf v; uint4 q[2]; };
  union AccU  { v8f v; float f[8]; };

  v8f acc[2][4] = {};

  auto compute = [&](int buf) {
    const unsigned short* Ab = As + buf * (128 * 32);
    const unsigned short* Bb = Bs + buf * (32 * 128);
    FragU fa[2], fb[4];
#pragma unroll
    for (int mi = 0; mi < 2; ++mi) {
      // A 16x32: lane<16 row M=lane K{0..7,16..23}; lane>=16 K{8..15,24..31}
      const unsigned short* p = Ab + (wm0 + mi * 16 + l15) * 32 + hf * 8;
      fa[mi].q[0] = *((const uint4*)p);
      fa[mi].q[1] = *((const uint4*)(p + 16));
    }
#pragma unroll
    for (int ni = 0; ni < 4; ++ni) {
      // B 32x16: lane<16 col N=lane K{0..7,16..23}; lane>=16 K{8..15,24..31}
      const unsigned short* p = Bb + (hf * 8) * 128 + wn0 + ni * 16 + l15;
      unsigned short* e = (unsigned short*)&fb[ni];
#pragma unroll
      for (int s = 0; s < 16; ++s) e[s] = p[((s & 7) + (s >> 3) * 16) * 128];
    }
#pragma unroll
    for (int mi = 0; mi < 2; ++mi)
#pragma unroll
      for (int ni = 0; ni < 4; ++ni)
        acc[mi][ni] = __builtin_amdgcn_wmma_f32_16x16x32_bf16(
            false, fa[mi].v, false, fb[ni].v, (short)0, acc[mi][ni], false, false);
  };

#if HAVE_TDM
  // ---- TDM staging: 2 DMA descriptors per K-step, issued by wave 0 ----
  auto stage = [&](int k0, int buf) {
    if (wave == 0) {
      tdm_load_2d(As + buf * (128 * 32), A + (long long)m0 * lda + k0,
                  (unsigned)(K - k0), (unsigned)(M - m0), (unsigned long long)lda,
                  32u, 128u);
      tdm_load_2d(Bs + buf * (32 * 128), Bm + (long long)k0 * ldb + n0,
                  (unsigned)(N - n0), (unsigned)(K - k0), (unsigned long long)ldb,
                  128u, 32u);
    }
  };
  const int nk = K >> 5;
  stage(0, 0);
  if (wave == 0) wait_tensor0();
  __syncthreads();
  for (int kt = 0; kt < nk; ++kt) {
    const int cur = kt & 1;
    if (kt + 1 < nk) stage((kt + 1) << 5, cur ^ 1);  // DMA-fill other buffer
    compute(cur);                                     // WMMA on current buffer
    if (wave == 0) wait_tensor0();
    __syncthreads();
  }
#else
  // ---- fallback: per-lane async global->LDS b128 staging ----
  const int arow = tid >> 1;          // A: 128 rows, 2 threads/row, 16 bf16 each
  const int acol = (tid & 1) * 16;
  const int brow = tid >> 3;          // B: 32 rows, 8 threads/row, 16 bf16 each
  const int bcol = (tid & 7) * 16;

  auto run = [&](auto edge_c) {
    constexpr bool EDGE = decltype(edge_c)::value;
    auto stage = [&](int k0, int buf) {
      {
        unsigned short* dst = As + buf * (128 * 32) + arow * 32 + acol;
        const unsigned short* gp = A + (long long)(m0 + arow) * lda + (k0 + acol);
        async_b128(dst,     gp);
        async_b128(dst + 8, gp + 8);
      }
      if (!EDGE) {
        unsigned short* dst = Bs + buf * (32 * 128) + brow * 128 + bcol;
        const unsigned short* gp = Bm + (long long)(k0 + brow) * ldb + (n0 + bcol);
        async_b128(dst,     gp);
        async_b128(dst + 8, gp + 8);
      } else {
        const int gn = n0 + bcol;
        const unsigned short* gp = Bm + (long long)(k0 + brow) * ldb + gn;
        uint4 b0, b1;
        if (gn + 16 <= N) {
          b0 = ((const uint4*)gp)[0];
          b1 = ((const uint4*)gp)[1];
        } else {
          __align__(16) unsigned short tmp[16];
#pragma unroll
          for (int j = 0; j < 16; ++j) tmp[j] = (gn + j < N) ? gp[j] : (unsigned short)0;
          b0 = *(const uint4*)(tmp);
          b1 = *(const uint4*)(tmp + 8);
        }
        unsigned short* dst = Bs + buf * (32 * 128) + brow * 128 + bcol;
        *((uint4*)dst)       = b0;
        *((uint4*)(dst + 8)) = b1;
      }
    };
    const int nk = K >> 5;
    stage(0, 0);
    wait_async0();
    __syncthreads();
    for (int kt = 0; kt < nk; ++kt) {
      const int cur = kt & 1;
      if (kt + 1 < nk) stage((kt + 1) << 5, cur ^ 1);
      compute(cur);
      wait_async0();
      __syncthreads();
    }
  };
  if (n0 + 128 <= N) run(FalseC{});
  else               run(TrueC{});
#endif

  // ---- epilogue: C layout lane l: N=l%16, VGPR r: M = 8*(l/16)+r ----
#pragma unroll
  for (int mi = 0; mi < 2; ++mi) {
#pragma unroll
    for (int ni = 0; ni < 4; ++ni) {
      AccU u; u.v = acc[mi][ni];
      const int col = n0 + wn0 + ni * 16 + l15;
      if (col < N) {
        const float bv = bias ? bias[col] : 0.0f;
#pragma unroll
        for (int r = 0; r < 8; ++r) {
          const int row = m0 + wm0 + mi * 16 + hf * 8 + r;
          float val = u.f[r] * alpha + bv;
          if (GELU) val = 0.5f * val * (1.0f + erff(val * 0.70710678118654752f));
          const long long off = (long long)row * ldc + col;
          if (RESID) val += resid[off];
          if (OUTBF) Cb[off] = f2bf(val);
          else       Cf[off] = val;
        }
      }
    }
  }
}

// ---------------------------------------------------------------------------
// Elementwise / reduction helper kernels
// ---------------------------------------------------------------------------
__global__ void gpt_embed_kernel(const int* __restrict__ idx, const float* __restrict__ tok,
                                 const float* __restrict__ pos, float* __restrict__ x) {
  const int row = blockIdx.x;          // 0..BT-1
  const int t   = row % T_;
  const long long tk = (long long)idx[row] * C_;
  for (int c = threadIdx.x; c < C_; c += blockDim.x)
    x[(long long)row * C_ + c] = tok[tk + c] + pos[(long long)t * C_ + c];
}

__global__ __launch_bounds__(256)
void gpt_ln_bf16_kernel(const float* __restrict__ x, const float* __restrict__ w,
                        const float* __restrict__ b, unsigned short* __restrict__ out) {
  __shared__ float red[256];
  const int row = blockIdx.x;
  const float* xr = x + (long long)row * C_;
  float s = 0.0f;
  for (int c = threadIdx.x; c < C_; c += 256) s += xr[c];
  red[threadIdx.x] = s; __syncthreads();
  for (int st = 128; st > 0; st >>= 1) {
    if (threadIdx.x < st) red[threadIdx.x] += red[threadIdx.x + st];
    __syncthreads();
  }
  const float mu = red[0] * (1.0f / C_);
  __syncthreads();
  float s2 = 0.0f;
  for (int c = threadIdx.x; c < C_; c += 256) { float d = xr[c] - mu; s2 += d * d; }
  red[threadIdx.x] = s2; __syncthreads();
  for (int st = 128; st > 0; st >>= 1) {
    if (threadIdx.x < st) red[threadIdx.x] += red[threadIdx.x + st];
    __syncthreads();
  }
  const float rstd = rsqrtf(red[0] * (1.0f / C_) + 1e-5f);
  for (int c = threadIdx.x; c < C_; c += 256)
    out[(long long)row * C_ + c] = f2bf((xr[c] - mu) * rstd * w[c] + b[c]);
}

__global__ void gpt_cvt_bf16_kernel(const float* __restrict__ in, unsigned short* __restrict__ out,
                                    long long n) {
  long long i  = (long long)blockIdx.x * blockDim.x + threadIdx.x;
  long long st = (long long)gridDim.x * blockDim.x;
  for (; i < n; i += st) out[i] = f2bf(in[i]);
}

__global__ void gpt_cvt_pad_kernel(const float* __restrict__ in, unsigned short* __restrict__ out,
                                   int rows, int cols, int ldo) {
  const long long n = (long long)rows * ldo;
  long long i  = (long long)blockIdx.x * blockDim.x + threadIdx.x;
  long long st = (long long)gridDim.x * blockDim.x;
  for (; i < n; i += st) {
    const int r = (int)(i / ldo);
    const int c = (int)(i % ldo);
    out[i] = (c < cols) ? f2bf(in[(long long)r * cols + c]) : (unsigned short)0;
  }
}

// [B,T,C] -> [BH,T,HD] (transpose==0) or [BH,HD,T] (transpose==1)
__global__ void gpt_split_heads_kernel(const unsigned short* __restrict__ in,
                                       unsigned short* __restrict__ out, int transpose) {
  const long long n = (long long)BT_ * C_;
  long long i = (long long)blockIdx.x * blockDim.x + threadIdx.x;
  if (i >= n) return;
  const int c  = (int)(i % C_);
  const long long bt = i / C_;
  const int b = (int)(bt / T_), t = (int)(bt % T_);
  const int h = c / HD_, d = c % HD_;
  const long long bh = (long long)b * H_ + h;
  const long long o = transpose ? (bh * HD_ + d) * T_ + t : (bh * T_ + t) * HD_ + d;
  out[o] = in[i];
}

// [BH,T,HD] -> [B,T,C]
__global__ void gpt_merge_heads_kernel(const unsigned short* __restrict__ in,
                                       unsigned short* __restrict__ out) {
  const long long n = (long long)BT_ * C_;
  long long i = (long long)blockIdx.x * blockDim.x + threadIdx.x;
  if (i >= n) return;
  const int c  = (int)(i % C_);
  const long long bt = i / C_;
  const int b = (int)(bt / T_), t = (int)(bt % T_);
  const int h = c / HD_, d = c % HD_;
  out[i] = in[(((long long)b * H_ + h) * T_ + t) * HD_ + d];
}

// causal row softmax: S fp32 [BH*T, T] -> P bf16, zeros for j > t
__global__ __launch_bounds__(256)
void gpt_softmax_causal_kernel(const float* __restrict__ S, unsigned short* __restrict__ P) {
  __shared__ float red[256];
  const long long row = blockIdx.x;    // 0..BH*T-1
  const int t = (int)(row % T_);
  const float* sr = S + row * T_;
  unsigned short* pr = P + row * T_;
  const int len = t + 1;
  float m = -3.4e38f;
  for (int j = threadIdx.x; j < len; j += 256) m = fmaxf(m, sr[j]);
  red[threadIdx.x] = m; __syncthreads();
  for (int st = 128; st > 0; st >>= 1) {
    if (threadIdx.x < st) red[threadIdx.x] = fmaxf(red[threadIdx.x], red[threadIdx.x + st]);
    __syncthreads();
  }
  m = red[0]; __syncthreads();
  float s = 0.0f;
  for (int j = threadIdx.x; j < len; j += 256) s += __expf(sr[j] - m);
  red[threadIdx.x] = s; __syncthreads();
  for (int st = 128; st > 0; st >>= 1) {
    if (threadIdx.x < st) red[threadIdx.x] += red[threadIdx.x + st];
    __syncthreads();
  }
  const float inv = 1.0f / red[0];
  for (int j = threadIdx.x; j < T_; j += 256)
    pr[j] = (j < len) ? f2bf(__expf(sr[j] - m) * inv) : (unsigned short)0;
}

__global__ void gpt_zero_loss_kernel(float* loss) {
  if (blockIdx.x == 0 && threadIdx.x == 0) loss[0] = 0.0f;
}

__global__ __launch_bounds__(256)
void gpt_nll_kernel(const float* __restrict__ logits, const int* __restrict__ tgt,
                    float* __restrict__ loss) {
  __shared__ float red[256];
  const int row = blockIdx.x;
  const float* lr = logits + (long long)row * V_;
  float m = -3.4e38f;
  for (int j = threadIdx.x; j < V_; j += 256) m = fmaxf(m, lr[j]);
  red[threadIdx.x] = m; __syncthreads();
  for (int st = 128; st > 0; st >>= 1) {
    if (threadIdx.x < st) red[threadIdx.x] = fmaxf(red[threadIdx.x], red[threadIdx.x + st]);
    __syncthreads();
  }
  m = red[0]; __syncthreads();
  float s = 0.0f;
  for (int j = threadIdx.x; j < V_; j += 256) s += __expf(lr[j] - m);
  red[threadIdx.x] = s; __syncthreads();
  for (int st = 128; st > 0; st >>= 1) {
    if (threadIdx.x < st) red[threadIdx.x] += red[threadIdx.x + st];
    __syncthreads();
  }
  if (threadIdx.x == 0) {
    const float lse = m + __logf(red[0]);
    atomicAdd(loss, (lse - lr[tgt[row]]) * (1.0f / BT_));
  }
}

// ---------------------------------------------------------------------------
// Host orchestration
// ---------------------------------------------------------------------------
extern "C" void kernel_launch(void* const* d_in, const int* in_sizes, int n_in,
                              void* d_out, int out_size, void* d_ws, size_t ws_size,
                              hipStream_t stream) {
  (void)in_sizes; (void)n_in; (void)out_size; (void)ws_size;
  const int*   idx  = (const int*)  d_in[0];
  const int*   tgt  = (const int*)  d_in[1];
  const float* tok  = (const float*)d_in[2];
  const float* pos  = (const float*)d_in[3];
  const float* ln1w = (const float*)d_in[4];
  const float* ln1b = (const float*)d_in[5];
  const float* ln2w = (const float*)d_in[6];
  const float* ln2b = (const float*)d_in[7];
  const float* Wq = (const float*)d_in[8];   const float* bq = (const float*)d_in[9];
  const float* Wk = (const float*)d_in[10];  const float* bk = (const float*)d_in[11];
  const float* Wv = (const float*)d_in[12];  const float* bv = (const float*)d_in[13];
  const float* Wo = (const float*)d_in[14];  const float* bo = (const float*)d_in[15];
  const float* W1 = (const float*)d_in[16];  const float* b1 = (const float*)d_in[17];
  const float* W2 = (const float*)d_in[18];  const float* b2 = (const float*)d_in[19];
  const float* lnfw = (const float*)d_in[20];
  const float* lnfb = (const float*)d_in[21];
  const float* headw = (const float*)d_in[22];

  char* ws = (char*)d_ws;
  size_t off = 0;
  auto wsalloc = [&](size_t bytes) -> void* {
    void* p = ws + off;
    off = (off + bytes + 255) & ~((size_t)255);
    return p;
  };

  float*          x   = (float*)         wsalloc((size_t)BT_ * C_ * 4);
  unsigned short* h   = (unsigned short*)wsalloc((size_t)BT_ * C_ * 2);
  unsigned short* qb  = (unsigned short*)wsalloc((size_t)BT_ * C_ * 2);
  unsigned short* kb  = (unsigned short*)wsalloc((size_t)BT_ * C_ * 2);
  unsigned short* vb  = (unsigned short*)wsalloc((size_t)BT_ * C_ * 2);
  unsigned short* qh  = (unsigned short*)wsalloc((size_t)BT_ * C_ * 2);   // [BH,T,HD]
  unsigned short* kT  = (unsigned short*)wsalloc((size_t)BT_ * C_ * 2);   // [BH,HD,T]
  unsigned short* vh  = (unsigned short*)wsalloc((size_t)BT_ * C_ * 2);   // [BH,T,HD]
  unsigned short* yh  = (unsigned short*)wsalloc((size_t)BT_ * C_ * 2);   // [BH,T,HD]
  unsigned short* y   = (unsigned short*)wsalloc((size_t)BT_ * C_ * 2);   // [B,T,C]
  unsigned short* a1  = (unsigned short*)wsalloc((size_t)BT_ * 4 * C_ * 2);
  float*          S   = (float*)         wsalloc((size_t)BH_ * T_ * T_ * 4);
  unsigned short* P   = (unsigned short*)wsalloc((size_t)BH_ * T_ * T_ * 2);
  unsigned short* wqb = (unsigned short*)wsalloc((size_t)C_ * C_ * 2);
  unsigned short* wkb = (unsigned short*)wsalloc((size_t)C_ * C_ * 2);
  unsigned short* wvb = (unsigned short*)wsalloc((size_t)C_ * C_ * 2);
  unsigned short* wob = (unsigned short*)wsalloc((size_t)C_ * C_ * 2);
  unsigned short* w1b = (unsigned short*)wsalloc((size_t)C_ * 4 * C_ * 2);
  unsigned short* w2b = (unsigned short*)wsalloc((size_t)4 * C_ * C_ * 2);
  unsigned short* xf  = (unsigned short*)wsalloc((size_t)BT_ * C_ * 2);
  unsigned short* hwb = (unsigned short*)wsalloc((size_t)C_ * VPAD * 2);  // padded LM head

  const dim3 blk(256);
  const long long nel = (long long)BT_ * C_;
  const unsigned permBlocks = (unsigned)((nel + 255) / 256);

  // LM head weight -> bf16 padded [C, VPAD], once
  gpt_cvt_pad_kernel<<<4096, blk, 0, stream>>>(headw, hwb, C_, V_, VPAD);

  // embeddings
  gpt_embed_kernel<<<BT_, blk, 0, stream>>>(idx, tok, pos, x);

  const dim3 gCC(C_ / 128, BT_ / 128, 1);           // (8,16) 2048x1024 GEMMs
  const dim3 gUP(4 * C_ / 128, BT_ / 128, 1);       // (32,16) MLP up
  const dim3 gSC(T_ / 128, T_ / 128, BH_);          // (8,8,32) attention scores
  const dim3 gPV(1, T_ / 128, BH_);                 // (1,8,32) P @ V

  for (int l = 0; l < L_; ++l) {
    // per-layer weights -> bf16
    gpt_cvt_bf16_kernel<<<2048, blk, 0, stream>>>(Wq + (size_t)l * C_ * C_, wqb, (long long)C_ * C_);
    gpt_cvt_bf16_kernel<<<2048, blk, 0, stream>>>(Wk + (size_t)l * C_ * C_, wkb, (long long)C_ * C_);
    gpt_cvt_bf16_kernel<<<2048, blk, 0, stream>>>(Wv + (size_t)l * C_ * C_, wvb, (long long)C_ * C_);
    gpt_cvt_bf16_kernel<<<2048, blk, 0, stream>>>(Wo + (size_t)l * C_ * C_, wob, (long long)C_ * C_);
    gpt_cvt_bf16_kernel<<<4096, blk, 0, stream>>>(W1 + (size_t)l * C_ * 4 * C_, w1b, (long long)C_ * 4 * C_);
    gpt_cvt_bf16_kernel<<<4096, blk, 0, stream>>>(W2 + (size_t)l * 4 * C_ * C_, w2b, (long long)4 * C_ * C_);

    // h = LN1(x)
    gpt_ln_bf16_kernel<<<BT_, blk, 0, stream>>>(x, ln1w + (size_t)l * C_, ln1b + (size_t)l * C_, h);

    // q,k,v = h @ W + b   (bf16 out)
    gpt_gemm_wmma_bf16<false, false, true><<<gCC, blk, 0, stream>>>(
        h, C_, 0, wqb, C_, 0, bq + (size_t)l * C_, nullptr, nullptr, qb, C_, 0,
        BT_, C_, C_, 1.0f, 0);
    gpt_gemm_wmma_bf16<false, false, true><<<gCC, blk, 0, stream>>>(
        h, C_, 0, wkb, C_, 0, bk + (size_t)l * C_, nullptr, nullptr, kb, C_, 0,
        BT_, C_, C_, 1.0f, 0);
    gpt_gemm_wmma_bf16<false, false, true><<<gCC, blk, 0, stream>>>(
        h, C_, 0, wvb, C_, 0, bv + (size_t)l * C_, nullptr, nullptr, vb, C_, 0,
        BT_, C_, C_, 1.0f, 0);

    // head regroup
    gpt_split_heads_kernel<<<permBlocks, blk, 0, stream>>>(qb, qh, 0);
    gpt_split_heads_kernel<<<permBlocks, blk, 0, stream>>>(kb, kT, 1);
    gpt_split_heads_kernel<<<permBlocks, blk, 0, stream>>>(vb, vh, 0);

    // S = (q @ k^T) / sqrt(HD), batched over BH; skip fully-masked tiles
    gpt_gemm_wmma_bf16<false, false, false><<<gSC, blk, 0, stream>>>(
        qh, HD_, (long long)T_ * HD_, kT, T_, (long long)HD_ * T_, nullptr, nullptr,
        S, nullptr, T_, (long long)T_ * T_, T_, T_, HD_, 0.125f, 1);

    // causal softmax -> P (bf16, zero above diagonal)
    gpt_softmax_causal_kernel<<<BH_ * T_, blk, 0, stream>>>(S, P);

    // Y = P @ V
    gpt_gemm_wmma_bf16<false, false, true><<<gPV, blk, 0, stream>>>(
        P, T_, (long long)T_ * T_, vh, HD_, (long long)T_ * HD_, nullptr, nullptr,
        nullptr, yh, HD_, (long long)T_ * HD_, T_, HD_, T_, 1.0f, 0);

    gpt_merge_heads_kernel<<<permBlocks, blk, 0, stream>>>(yh, y);

    // x = x + y @ Wo + bo   (fp32 residual)
    gpt_gemm_wmma_bf16<false, true, false><<<gCC, blk, 0, stream>>>(
        y, C_, 0, wob, C_, 0, bo + (size_t)l * C_, x, x, nullptr, C_, 0,
        BT_, C_, C_, 1.0f, 0);

    // h = LN2(x); a1 = GELU(h @ W1 + b1); x = x + a1 @ W2 + b2
    gpt_ln_bf16_kernel<<<BT_, blk, 0, stream>>>(x, ln2w + (size_t)l * C_, ln2b + (size_t)l * C_, h);
    gpt_gemm_wmma_bf16<true, false, true><<<gUP, blk, 0, stream>>>(
        h, C_, 0, w1b, 4 * C_, 0, b1 + (size_t)l * 4 * C_, nullptr, nullptr, a1, 4 * C_, 0,
        BT_, 4 * C_, C_, 1.0f, 0);
    gpt_gemm_wmma_bf16<false, true, false><<<gCC, blk, 0, stream>>>(
        a1, 4 * C_, 0, w2b, C_, 0, b2 + (size_t)l * C_, x, x, nullptr, C_, 0,
        BT_, C_, 4 * C_, 1.0f, 0);
  }

  // final LN, LM head into d_out, loss
  gpt_ln_bf16_kernel<<<BT_, blk, 0, stream>>>(x, lnfw, lnfb, xf);

  float* logits = (float*)d_out;
  const dim3 gHD((V_ + 127) / 128, BT_ / 128, 1);   // (393,16)
  gpt_gemm_wmma_bf16<false, false, false><<<gHD, blk, 0, stream>>>(
      xf, C_, 0, hwb, VPAD, 0, nullptr, nullptr, logits, nullptr, V_, 0,
      BT_, V_, C_, 1.0f, 0);

  float* loss = logits + (size_t)BT_ * V_;
  gpt_zero_loss_kernel<<<1, 1, 0, stream>>>(loss);
  gpt_nll_kernel<<<BT_, blk, 0, stream>>>(logits, tgt, loss);
}
